// GATModelVAE_4887672783331
// MI455X (gfx1250) — matrix operand order repaired and checked by reading, not compiled, not run on hardware
//
#include <hip/hip_runtime.h>

typedef __attribute__((ext_vector_type(16))) __bf16 v16bf;
typedef __attribute__((ext_vector_type(8)))  float  v8f;

#define HEADS 5
#define NEG_SLOPE 0.2f

__device__ __forceinline__ unsigned short f2bf(float f) {
  unsigned u = __float_as_uint(f);
  u += 0x7fffu + ((u >> 16) & 1u);   // round-to-nearest-even
  return (unsigned short)(u >> 16);
}

union BF16x16 { v16bf v; uint4 q[2]; unsigned short h[16]; };

__device__ __forceinline__ unsigned map_f(float v) {
  unsigned u = __float_as_uint(v);
  return (u & 0x80000000u) ? ~u : (u | 0x80000000u);
}
__device__ __forceinline__ float unmap_f(unsigned m) {
  unsigned u = (m & 0x80000000u) ? (m & 0x7fffffffu) : ~m;
  return __uint_as_float(u);
}

// f32 -> bf16 (row-major copy); n must be a multiple of 8
__global__ void k_cvt_bf16(const float* __restrict__ in,
                           unsigned short* __restrict__ out, int n) {
  int base = (blockIdx.x * blockDim.x + threadIdx.x) * 8;
  if (base >= n) return;
  unsigned short tmp[8];
#pragma unroll
  for (int j = 0; j < 8; ++j) tmp[j] = f2bf(in[base + j]);
  *(uint4*)(out + base) = *(const uint4*)tmp;
}

// f32 [K,Nn] -> bf16 transposed [Nn,K]
__global__ void k_cvt_bf16_T(const float* __restrict__ in,
                             unsigned short* __restrict__ out, int K, int Nn) {
  int i = blockIdx.x * blockDim.x + threadIdx.x;
  if (i >= K * Nn) return;
  int nrow = i / K, k = i % K;
  out[i] = f2bf(in[(size_t)k * Nn + nrow]);
}

// C[M,Nn] = A @ B with A bf16 [M,K] row-major, BT bf16 [Nn,K] (B transposed).
// One wave computes a 32x32 C tile: 2x2 register blocking of 16x16 WMMA tiles,
// 4 v_wmma per K-step of 32, all operand fetches are contiguous b128 loads.
// Tile id is wave-uniform -> EXEC all-1s at every WMMA (ISA requirement).
__global__ void k_gemm_wmma_bf16(const unsigned short* __restrict__ A,
                                 const unsigned short* __restrict__ BT,
                                 float* __restrict__ C, int M, int Nn, int K) {
  int lane  = threadIdx.x & 31;
  int wave  = threadIdx.x >> 5;
  int tile  = blockIdx.x * (blockDim.x >> 5) + wave;
  int ntn   = Nn >> 5;
  int ntile = (M >> 5) * ntn;
  if (tile >= ntile) return;
  int mr = (tile / ntn) << 5;
  int nc = (tile % ntn) << 5;
  int row   = lane & 15;   // A row / B col / C col within 16x16 sub-tile
  int khalf = lane >> 4;

  const unsigned short* a0p = A  + (size_t)(mr + row)      * K;
  const unsigned short* a1p = A  + (size_t)(mr + 16 + row) * K;
  const unsigned short* b0p = BT + (size_t)(nc + row)      * K;
  const unsigned short* b1p = BT + (size_t)(nc + 16 + row) * K;

  v8f acc00 = {}, acc01 = {}, acc10 = {}, acc11 = {};
  for (int kb = 0; kb < K; kb += 32) {
    BF16x16 a0, a1, b0, b1;
    // A 16x32 bf16 frag: h[0..7] = K kb+khalf*8+0..7 ; h[8..15] = +16
    a0.q[0] = *(const uint4*)(a0p + kb + khalf * 8);
    a0.q[1] = *(const uint4*)(a0p + kb + 16 + khalf * 8);
    a1.q[0] = *(const uint4*)(a1p + kb + khalf * 8);
    a1.q[1] = *(const uint4*)(a1p + kb + 16 + khalf * 8);
    // B 32x16 bf16 frag: h[0..7] = K kb+khalf*8+0..7 ; h[8..15] = +16 (same as A)
    b0.q[0] = *(const uint4*)(b0p + kb + khalf * 8);
    b0.q[1] = *(const uint4*)(b0p + kb + 16 + khalf * 8);
    b1.q[0] = *(const uint4*)(b1p + kb + khalf * 8);
    b1.q[1] = *(const uint4*)(b1p + kb + 16 + khalf * 8);
    if (kb + 32 < K) {                       // gfx1250 global_prefetch_b8 path
      __builtin_prefetch(a0p + kb + 32, 0, 0);
      __builtin_prefetch(a1p + kb + 32, 0, 0);
    }
    acc00 = __builtin_amdgcn_wmma_f32_16x16x32_bf16(false, a0.v, false, b0.v,
                                                    (short)0, acc00, false, false);
    acc01 = __builtin_amdgcn_wmma_f32_16x16x32_bf16(false, a0.v, false, b1.v,
                                                    (short)0, acc01, false, false);
    acc10 = __builtin_amdgcn_wmma_f32_16x16x32_bf16(false, a1.v, false, b0.v,
                                                    (short)0, acc10, false, false);
    acc11 = __builtin_amdgcn_wmma_f32_16x16x32_bf16(false, a1.v, false, b1.v,
                                                    (short)0, acc11, false, false);
  }
  // C f32 16x16 layout: VGPR r -> M = r + khalf*8, lane&15 -> N
#pragma unroll
  for (int r = 0; r < 8; ++r) {
    int m0 = mr + r + khalf * 8;
    int m1 = m0 + 16;
    C[(size_t)m0 * Nn + nc + row]      = acc00[r];
    C[(size_t)m0 * Nn + nc + 16 + row] = acc01[r];
    C[(size_t)m1 * Nn + nc + row]      = acc10[r];
    C[(size_t)m1 * Nn + nc + 16 + row] = acc11[r];
  }
}

__global__ void k_fill_u32(unsigned* __restrict__ p, unsigned val, int n) {
  int i = blockIdx.x * blockDim.x + threadIdx.x;
  if (i < n) p[i] = val;
}

// a_s[n,h] = sum_c xw[n,h,c]*att_src[h,c]; a_d likewise. One thread per (n,h).
__global__ void k_attn_scores(const float* __restrict__ xw,
                              const float* __restrict__ att_src,
                              const float* __restrict__ att_dst,
                              float* __restrict__ a_s, float* __restrict__ a_d,
                              int N, int C) {
  int i = blockIdx.x * blockDim.x + threadIdx.x;
  if (i >= N * HEADS) return;
  int h = i % HEADS;
  const float* xp = xw + (size_t)i * C;
  const float* as = att_src + h * C;
  const float* ad = att_dst + h * C;
  float s = 0.f, d = 0.f;
  for (int c = 0; c < C; ++c) { float x = xp[c]; s += x * as[c]; d += x * ad[c]; }
  a_s[i] = s;
  a_d[i] = d;
}

__device__ __forceinline__ void edge_sd(const int* __restrict__ ei, int E, int e,
                                        int& s, int& d) {
  if (e < E) { s = ei[e]; d = ei[E + e]; }
  else       { s = e - E; d = s; }        // self loops appended
}

// scatter-max of leaky_relu(a_s[src]+a_d[dst]) into m[dst,h] (order-mapped uint)
__global__ void k_edge_max(const int* __restrict__ ei, int E, int N,
                           const float* __restrict__ a_s, const float* __restrict__ a_d,
                           unsigned* __restrict__ mmax) {
  int e = blockIdx.x * blockDim.x + threadIdx.x;
  if (e >= E + N) return;
  int s, d; edge_sd(ei, E, e, s, d);
#pragma unroll
  for (int h = 0; h < HEADS; ++h) {
    float v = a_s[s * HEADS + h] + a_d[d * HEADS + h];
    v = v > 0.f ? v : NEG_SLOPE * v;
    atomicMax(&mmax[d * HEADS + h], map_f(v));
  }
}

// alpha_unnorm[e,h] = exp(e - m[dst]); denom[dst,h] += alpha_unnorm
__global__ void k_edge_expsum(const int* __restrict__ ei, int E, int N,
                              const float* __restrict__ a_s, const float* __restrict__ a_d,
                              const unsigned* __restrict__ mmax,
                              float* __restrict__ denom, float* __restrict__ alpha) {
  int e = blockIdx.x * blockDim.x + threadIdx.x;
  if (e >= E + N) return;
  int s, d; edge_sd(ei, E, e, s, d);
#pragma unroll
  for (int h = 0; h < HEADS; ++h) {
    float v = a_s[s * HEADS + h] + a_d[d * HEADS + h];
    v = v > 0.f ? v : NEG_SLOPE * v;
    float ex = __expf(v - unmap_f(mmax[d * HEADS + h]));
    alpha[(size_t)e * HEADS + h] = ex;
    atomicAdd(&denom[d * HEADS + h], ex);
  }
}

// agg[dst,h,c] += (alpha/denom) * xw[src,h,c]; one thread per (edge, c)
__global__ void k_edge_agg(const int* __restrict__ ei, int E, int N, int C,
                           const float* __restrict__ xw, const float* __restrict__ alpha,
                           const float* __restrict__ denom, float* __restrict__ agg) {
  int t = blockIdx.x * blockDim.x + threadIdx.x;
  int ET = E + N;
  if (t >= ET * C) return;
  int e = t / C, c = t % C;
  int s, d; edge_sd(ei, E, e, s, d);
#pragma unroll
  for (int h = 0; h < HEADS; ++h) {
    float w = alpha[(size_t)e * HEADS + h] / (denom[d * HEADS + h] + 1e-16f);
    atomicAdd(&agg[((size_t)d * HEADS + h) * C + c],
              w * xw[((size_t)s * HEADS + h) * C + c]);
  }
}

// out[n,c] = act(mean_h agg[n,h,c] + b[c]); optional second output (z = mu)
__global__ void k_head_mean(const float* __restrict__ agg, const float* __restrict__ bias,
                            float* __restrict__ out, float* __restrict__ out2,
                            int N, int C, int do_relu) {
  int i = blockIdx.x * blockDim.x + threadIdx.x;
  if (i >= N * C) return;
  int n = i / C, c = i % C;
  float s = 0.f;
#pragma unroll
  for (int h = 0; h < HEADS; ++h) s += agg[((size_t)n * HEADS + h) * C + c];
  float o = s * (1.0f / HEADS) + bias[c];
  if (do_relu) o = fmaxf(o, 0.f);
  out[i] = o;
  if (out2) out2[i] = o;
}

extern "C" void kernel_launch(void* const* d_in, const int* in_sizes, int n_in,
                              void* d_out, int out_size, void* d_ws, size_t ws_size,
                              hipStream_t stream) {
  (void)in_sizes; (void)n_in; (void)out_size; (void)ws_size;
  const int N = 20000, E = 320000;

  const float* x   = (const float*)d_in[0];
  const int*   ei  = (const int*)d_in[1];
  const float* W1  = (const float*)d_in[2];
  const float* as1 = (const float*)d_in[3];
  const float* ad1 = (const float*)d_in[4];
  const float* b1  = (const float*)d_in[5];
  const float* W2  = (const float*)d_in[6];
  const float* as2 = (const float*)d_in[7];
  const float* ad2 = (const float*)d_in[8];
  const float* b2  = (const float*)d_in[9];
  const float* W3  = (const float*)d_in[10];
  const float* as3 = (const float*)d_in[11];
  const float* ad3 = (const float*)d_in[12];
  const float* b3  = (const float*)d_in[13];
  float* out = (float*)d_out;

  // workspace layout (float units; every region even-sized -> 16B alignment kept)
  float* ws      = (float*)d_ws;
  float* f_xw1   = ws;                                    // N*320 (reused xw2|xw3)
  float* f_agg   = f_xw1   + (size_t)N * 320;             // N*320
  float* f_as    = f_agg   + (size_t)N * 320;             // N*H
  float* f_ad    = f_as    + (size_t)N * HEADS;           // N*H
  unsigned* u_m  = (unsigned*)(f_ad + (size_t)N * HEADS); // N*H
  float* f_den   = (float*)(u_m + (size_t)N * HEADS);     // N*H
  float* f_alpha = f_den   + (size_t)N * HEADS;           // (E+N)*H
  float* f_h1    = f_alpha + (size_t)(E + N) * HEADS;     // N*64
  unsigned short* a16  = (unsigned short*)(f_h1 + (size_t)N * 64); // N*256 bf16
  unsigned short* bt1  = a16 + (size_t)N * 256;           // 320*256 bf16
  unsigned short* bt2  = bt1 + (size_t)320 * 256;         // 160*64 bf16
  unsigned short* bt3  = bt2 + (size_t)160 * 64;          // 160*64 bf16
  float* f_xw2   = f_xw1;
  float* f_xw3   = f_xw1 + (size_t)N * 160;

  auto launch_gemm = [&](const unsigned short* A, const unsigned short* BT, float* Cc,
                         int M, int Nn, int K) {
    int tiles  = (M / 32) * (Nn / 32);
    int wpb    = 8;                               // 8 waves = 256 threads
    int blocks = (tiles + wpb - 1) / wpb;
    k_gemm_wmma_bf16<<<blocks, wpb * 32, 0, stream>>>(A, BT, Cc, M, Nn, K);
  };

  auto run_layer = [&](const float* xw, int C, const float* attS, const float* attD,
                       const float* bias, float* o1, float* o2, int relu) {
    int NH = N * HEADS;
    int ET = E + N;
    k_fill_u32<<<(NH + 255) / 256, 256, 0, stream>>>(u_m, 0u, NH);
    k_fill_u32<<<(NH + 255) / 256, 256, 0, stream>>>((unsigned*)f_den, 0u, NH);
    int aggN = N * HEADS * C;
    k_fill_u32<<<(aggN + 255) / 256, 256, 0, stream>>>((unsigned*)f_agg, 0u, aggN);
    k_attn_scores<<<(NH + 255) / 256, 256, 0, stream>>>(xw, attS, attD, f_as, f_ad, N, C);
    k_edge_max<<<(ET + 255) / 256, 256, 0, stream>>>(ei, E, N, f_as, f_ad, u_m);
    k_edge_expsum<<<(ET + 255) / 256, 256, 0, stream>>>(ei, E, N, f_as, f_ad, u_m,
                                                        f_den, f_alpha);
    int tot = ET * C;
    k_edge_agg<<<(tot + 255) / 256, 256, 0, stream>>>(ei, E, N, C, xw, f_alpha,
                                                      f_den, f_agg);
    k_head_mean<<<(N * C + 255) / 256, 256, 0, stream>>>(f_agg, bias, o1, o2, N, C, relu);
  };

  // ---- Layer 1: xw1 = x @ W1 ; GAT ; relu -> h1 ----
  {
    int n = N * 256;
    k_cvt_bf16<<<(n / 8 + 255) / 256, 256, 0, stream>>>(x, a16, n);
    int t = 256 * 320;
    k_cvt_bf16_T<<<(t + 255) / 256, 256, 0, stream>>>(W1, bt1, 256, 320);
    launch_gemm(a16, bt1, f_xw1, N, 320, 256);
  }
  run_layer(f_xw1, 64, as1, ad1, b1, f_h1, nullptr, 1);

  // ---- Layers 2 & 3: shared h1 -> bf16 once ----
  {
    int n = N * 64;
    k_cvt_bf16<<<(n / 8 + 255) / 256, 256, 0, stream>>>(f_h1, a16, n);
    int t = 64 * 160;
    k_cvt_bf16_T<<<(t + 255) / 256, 256, 0, stream>>>(W2, bt2, 64, 160);
    k_cvt_bf16_T<<<(t + 255) / 256, 256, 0, stream>>>(W3, bt3, 64, 160);
    launch_gemm(a16, bt2, f_xw2, N, 160, 64);
    launch_gemm(a16, bt3, f_xw3, N, 160, 64);
  }
  // mu (also written into z slot: eval-mode z = mu), then logvar
  run_layer(f_xw2, 32, as2, ad2, b2, out + (size_t)N * 32, out, 0);
  run_layer(f_xw3, 32, as3, ad3, b3, out + (size_t)2 * N * 32, nullptr, 0);
}
